// GAT0tmaster_33200097198299
// MI455X (gfx1250) — compile-verified, matrix operand order checked by reading, not run on hardware
//
#include <hip/hip_runtime.h>
#include <hip/hip_bf16.h>
#include <math.h>

// ---------------- problem constants (match reference) ----------------
#define NN 10000      // N_NODES
#define NE 160000     // N_EDGES
#define NG 16         // N_GRAPHS
#define HEADS 4
#define NEG_SLOPE 0.2f
#define TK 64         // K-chunk staged in LDS per iteration (2 WMMA k-steps)

typedef __attribute__((ext_vector_type(16))) __bf16 v16bf;
typedef __attribute__((ext_vector_type(8)))  float  v8f;

// f32 -> bf16 (round-to-nearest-even), raw bit form for LDS staging
__device__ __forceinline__ unsigned short f2bfu(float f) {
  unsigned u = __builtin_bit_cast(unsigned, f);
  return (unsigned short)((u + 0x7FFFu + ((u >> 16) & 1u)) >> 16);
}

// float atomic-max via signed/unsigned integer atomics (monotone bit trick)
__device__ __forceinline__ void atomicMaxF(float* addr, float val) {
  if (val >= 0.0f)
    atomicMax((int*)addr, __float_as_int(val));
  else
    atomicMin((unsigned int*)addr, (unsigned int)__float_as_int(val));
}

// ---------------- generic fill ----------------
__global__ void fill_kernel(float* __restrict__ p, float v, size_t n) {
  size_t i = (size_t)blockIdx.x * blockDim.x + threadIdx.x;
  if (i < n) p[i] = v;
}

// ---------------- bf16 WMMA GEMM (LDS-staged):  Y[N,M] = X[N,K] @ W[K,M] + bias[M] ----
// Block = 256 threads (8 waves) -> 16 rows x 128 cols of Y. Requires:
//   N % 16 == 0 (10000 ok), K % TK == 0 (256/1024 ok), M % 128 == 0 (1024/256 ok).
// All waves always active -> __syncthreads() is safe and EXEC is all-ones at WMMA.
__global__ void wmma_gemm_bias(const float* __restrict__ X, const float* __restrict__ W,
                               const float* __restrict__ bias, float* __restrict__ Y,
                               int K, int M) {
  __shared__ unsigned short lA[16 * TK];    // A tile  [row][k], bf16 bits
  __shared__ unsigned short lB[128 * TK];   // W tile transposed [colLocal][k], bf16 bits

  const int tid  = threadIdx.x;
  const int wave = tid >> 5;
  const int lane = tid & 31;
  const int r0   = blockIdx.x * 16;
  const int c0   = blockIdx.y * 128;
  const int rowl = lane & 15;               // fragment row (A) / col (B) within tile
  const int koff = (lane < 16) ? 0 : 8;     // ISA 16-bit fragment K offset for hi lanes
  const int cloc = wave * 16 + rowl;        // local column this lane owns in B/D

  v8f acc = {};
  for (int k0 = 0; k0 < K; k0 += TK) {
    // ---- stage X tile 16 x TK (1024 f32; 1 float4 per thread, coalesced) ----
    {
      int idx = tid * 4;                    // 0..1023
      int r = idx >> 6, kk = idx & 63;
      const float4 v = *(const float4*)(X + (size_t)(r0 + r) * K + (k0 + kk));
      unsigned short* dp = lA + r * TK + kk;
      dp[0] = f2bfu(v.x); dp[1] = f2bfu(v.y); dp[2] = f2bfu(v.z); dp[3] = f2bfu(v.w);
      if (k0 + TK < K)  // prefetch next X tile (global_prefetch_b8)
        __builtin_prefetch(X + (size_t)(r0 + r) * K + (k0 + TK + kk), 0, 1);
    }
    // ---- stage W tile TK x 128, transposed into LDS (8192 f32; 8 float4/thread) ----
#pragma unroll
    for (int i = 0; i < 8; ++i) {
      int idx = i * 1024 + tid * 4;         // 0..8191
      int k = idx >> 7, c = idx & 127;
      const float4 v = *(const float4*)(W + (size_t)(k0 + k) * M + (c0 + c));
      lB[(c + 0) * TK + k] = f2bfu(v.x);
      lB[(c + 1) * TK + k] = f2bfu(v.y);
      lB[(c + 2) * TK + k] = f2bfu(v.z);
      lB[(c + 3) * TK + k] = f2bfu(v.w);
      if (i == 0 && k0 + TK < K)            // prefetch next W tile
        __builtin_prefetch(W + (size_t)(k0 + TK + k) * M + (c0 + c), 0, 1);
    }
    __syncthreads();
    // ---- two 16x16x32 WMMA steps out of LDS (ds_load_b128 fragment fetches) ----
#pragma unroll
    for (int ks = 0; ks < TK; ks += 32) {
      struct U { uint4 lo, hi; } ua, ub;
      ua.lo = *(const uint4*)(lA + rowl * TK + ks + koff);        // K = ks+koff+0..7
      ua.hi = *(const uint4*)(lA + rowl * TK + ks + koff + 16);   // K = ks+koff+16..23
      ub.lo = *(const uint4*)(lB + cloc * TK + ks + koff);
      ub.hi = *(const uint4*)(lB + cloc * TK + ks + koff + 16);
      v16bf a = __builtin_bit_cast(v16bf, ua);
      v16bf b = __builtin_bit_cast(v16bf, ub);
      acc = __builtin_amdgcn_wmma_f32_16x16x32_bf16(false, a, false, b, (short)0, acc,
                                                    false, false);
    }
    __syncthreads();
  }
  // D layout: VGPR r -> row r + (lane<16?0:8), col = lane&15 (within wave's tile)
  const int col = c0 + cloc;
  const int rb  = r0 + ((lane < 16) ? 0 : 8);
  const float bv = bias[col];
#pragma unroll
  for (int r = 0; r < 8; ++r)
    Y[(size_t)(rb + r) * M + col] = acc[r] + bv;
}

// ---------------- mean edge-attr per destination node (fill_value='mean') ----------------
__global__ void edge_mean_acc(const int* __restrict__ dst, const float* __restrict__ ea,
                              float* __restrict__ mean_sum, float* __restrict__ deg) {
  int t = blockIdx.x * blockDim.x + threadIdx.x;
  if (t >= NE * 16) return;
  int e = t >> 4, k = t & 15;
  int d = dst[e];
  atomicAdd(mean_sum + (size_t)d * 16 + k, ea[t]);
  if (k == 0) atomicAdd(deg + d, 1.0f);
}
__global__ void edge_mean_div(float* __restrict__ mean_sum, const float* __restrict__ deg) {
  int t = blockIdx.x * blockDim.x + threadIdx.x;
  if (t >= NN * 16) return;
  mean_sum[t] /= fmaxf(deg[t >> 4], 1.0f);
}

// ---------------- attention logits + per-node running max ----------------
// one wave per (edge or self-loop) index e in [0, NE+NN)
template <int C>
__global__ void logits_kernel(const float* __restrict__ xl, const float* __restrict__ xr,
                              const int* __restrict__ src, const int* __restrict__ dst,
                              const float* __restrict__ edge_attr,
                              const float* __restrict__ mean_ea,
                              const float* __restrict__ We, const float* __restrict__ att,
                              float* __restrict__ logits, float* __restrict__ node_max) {
  const int HC = HEADS * C;
  int e = blockIdx.x * (blockDim.x >> 5) + (threadIdx.x >> 5);
  if (e >= NE + NN) return;
  const int lane = threadIdx.x & 31;
  const int s = (e < NE) ? src[e] : (e - NE);
  const int d = (e < NE) ? dst[e] : (e - NE);

  float ea[16];
  const bool has_edge = (edge_attr != nullptr);
  if (has_edge) {
    const float* eav = (e < NE) ? (edge_attr + (size_t)e * 16) : (mean_ea + (size_t)d * 16);
#pragma unroll
    for (int k = 0; k < 16; ++k) ea[k] = eav[k];
  }
#pragma unroll
  for (int h = 0; h < HEADS; ++h) {
    float acc = 0.0f;
    for (int cc = lane; cc < C; cc += 32) {
      int c = h * C + cc;
      float z = xl[(size_t)s * HC + c] + xr[(size_t)d * HC + c];
      if (has_edge) {
        float t = 0.0f;
#pragma unroll
        for (int k = 0; k < 16; ++k) t += ea[k] * We[(size_t)k * HC + c];
        z += t;
      }
      z = (z > 0.0f) ? z : NEG_SLOPE * z;    // leaky_relu
      acc += z * att[h * C + cc];
    }
#pragma unroll
    for (int off = 16; off; off >>= 1) acc += __shfl_xor(acc, off, 32);
    if (lane == 0) {
      logits[(size_t)e * HEADS + h] = acc;
      atomicMaxF(node_max + d * HEADS + h, acc);
    }
  }
}

// ---------------- exp / denominator / weighted scatter-add ----------------
template <int C>
__global__ void accum_kernel(const float* __restrict__ xl,
                             const int* __restrict__ src, const int* __restrict__ dst,
                             const float* __restrict__ logits,
                             const float* __restrict__ node_max,
                             float* __restrict__ node_sum, float* __restrict__ acc_out) {
  const int HC = HEADS * C;
  int e = blockIdx.x * (blockDim.x >> 5) + (threadIdx.x >> 5);
  if (e >= NE + NN) return;
  const int lane = threadIdx.x & 31;
  const int s = (e < NE) ? src[e] : (e - NE);
  const int d = (e < NE) ? dst[e] : (e - NE);

  float a[HEADS];
#pragma unroll
  for (int h = 0; h < HEADS; ++h)
    a[h] = expf(logits[(size_t)e * HEADS + h] - node_max[d * HEADS + h]);
  if (lane < HEADS) {
    float av = (lane == 0) ? a[0] : (lane == 1) ? a[1] : (lane == 2) ? a[2] : a[3];
    atomicAdd(node_sum + d * HEADS + lane, av);
  }
#pragma unroll
  for (int h = 0; h < HEADS; ++h) {
    float ah = a[h];
    for (int cc = h * C + lane; cc < (h + 1) * C; cc += 32)
      atomicAdd(acc_out + (size_t)d * HC + cc, xl[(size_t)s * HC + cc] * ah);
  }
}

// ---------------- normalize + bias (+ optional relu), in place ----------------
__global__ void final_kernel(float* __restrict__ out, const float* __restrict__ node_sum,
                             const float* __restrict__ bias, int C, int relu) {
  int HC = HEADS * C;
  size_t i = (size_t)blockIdx.x * blockDim.x + threadIdx.x;
  if (i >= (size_t)NN * HC) return;
  int n = (int)(i / HC), c = (int)(i % HC);
  float v = out[i] / node_sum[n * HEADS + c / C] + bias[c];
  if (relu) v = fmaxf(v, 0.0f);
  out[i] = v;
}

// ---------------- readout + MLP head: [16,256] -> relu fc1 -> fc2 -> [16,1] ----------------
__global__ void fc_kernel(const float* __restrict__ h3, const int* __restrict__ n_nodes,
                          const float* __restrict__ fc1_w, const float* __restrict__ fc1_b,
                          const float* __restrict__ fc2_w, const float* __restrict__ fc2_b,
                          float* __restrict__ out) {
  __shared__ float m[NG][64];
  int t = threadIdx.x;              // blockDim = 1024 = 16 graphs * 64 features
  int g = t >> 6, j = t & 63;
  int csum = 0;
  for (int i = 0; i <= g; ++i) csum += n_nodes[i];
  const float* hr = h3 + (size_t)(csum - 1) * 256;   // master node row
  float acc = fc1_b[j];
  for (int c = 0; c < 256; ++c) acc += hr[c] * fc1_w[c * 64 + j];
  m[g][j] = fmaxf(acc, 0.0f);
  __syncthreads();
  if (j == 0) {
    float o = fc2_b[0];
    for (int c = 0; c < 64; ++c) o += m[g][c] * fc2_w[c];
    out[g] = o;
  }
}

// ---------------- host-side layer driver ----------------
static void run_gatv2(hipStream_t st, const float* xin, int K, int C,
                      const float* Wl, const float* bl, const float* Wr, const float* br,
                      const float* We, const float* att, const float* bias,
                      const int* src, const int* dst,
                      const float* edge_attr, const float* mean_ea,
                      float* xl, float* xr, float* acc_out,
                      float* logits, float* node_max, float* node_sum, int relu) {
  const int HC = HEADS * C;
  dim3 gg(NN / 16, HC / 128);
  wmma_gemm_bias<<<gg, 256, 0, st>>>(xin, Wl, bl, xl, K, HC);
  wmma_gemm_bias<<<gg, 256, 0, st>>>(xin, Wr, br, xr, K, HC);

  fill_kernel<<<(NN * HEADS + 255) / 256, 256, 0, st>>>(node_max, -INFINITY, NN * HEADS);
  fill_kernel<<<(NN * HEADS + 255) / 256, 256, 0, st>>>(node_sum, 0.0f, NN * HEADS);
  {
    size_t n = (size_t)NN * HC;
    fill_kernel<<<(unsigned)((n + 255) / 256), 256, 0, st>>>(acc_out, 0.0f, n);
  }
  const int waves_blk = 8;                        // 256 threads
  const int eblocks = (NE + NN + waves_blk - 1) / waves_blk;
  if (C == 256) {
    logits_kernel<256><<<eblocks, 256, 0, st>>>(xl, xr, src, dst, edge_attr, mean_ea,
                                                We, att, logits, node_max);
    accum_kernel<256><<<eblocks, 256, 0, st>>>(xl, src, dst, logits, node_max,
                                               node_sum, acc_out);
  } else {
    logits_kernel<64><<<eblocks, 256, 0, st>>>(xl, xr, src, dst, edge_attr, mean_ea,
                                               We, att, logits, node_max);
    accum_kernel<64><<<eblocks, 256, 0, st>>>(xl, src, dst, logits, node_max,
                                              node_sum, acc_out);
  }
  size_t tot = (size_t)NN * HC;
  final_kernel<<<(unsigned)((tot + 255) / 256), 256, 0, st>>>(acc_out, node_sum, bias, C, relu);
}

extern "C" void kernel_launch(void* const* d_in, const int* in_sizes, int n_in,
                              void* d_out, int out_size, void* d_ws, size_t ws_size,
                              hipStream_t stream) {
  (void)in_sizes; (void)n_in; (void)out_size; (void)ws_size;
  // -------- inputs (top-level dict insertion order; params flattened with sorted keys) --------
  const float* x         = (const float*)d_in[0];            // [10000,256]
  const int*   eidx      = (const int*)d_in[1];              // [2,160000]
  const float* eattr     = (const float*)d_in[2];            // [160000,16]
  const int*   eidx_m    = (const int*)d_in[3];              // [2,160000]
  const int*   n_nodes   = (const int*)d_in[4];              // [16]
  // conv1: We, Wl, Wr, att, bias, bl, br
  const float* c1We  = (const float*)d_in[5];
  const float* c1Wl  = (const float*)d_in[6];
  const float* c1Wr  = (const float*)d_in[7];
  const float* c1att = (const float*)d_in[8];
  const float* c1b   = (const float*)d_in[9];
  const float* c1bl  = (const float*)d_in[10];
  const float* c1br  = (const float*)d_in[11];
  // conv2: We, Wl, Wr, att, bias, bl, br
  const float* c2We  = (const float*)d_in[12];
  const float* c2Wl  = (const float*)d_in[13];
  const float* c2Wr  = (const float*)d_in[14];
  const float* c2att = (const float*)d_in[15];
  const float* c2b   = (const float*)d_in[16];
  const float* c2bl  = (const float*)d_in[17];
  const float* c2br  = (const float*)d_in[18];
  // fc1_b, fc1_w, fc2_b, fc2_w
  const float* fc1_b = (const float*)d_in[19];
  const float* fc1_w = (const float*)d_in[20];
  const float* fc2_b = (const float*)d_in[21];
  const float* fc2_w = (const float*)d_in[22];
  // pool: Wl, Wr, att, bias, bl, br
  const float* pWl  = (const float*)d_in[23];
  const float* pWr  = (const float*)d_in[24];
  const float* patt = (const float*)d_in[25];
  const float* pb   = (const float*)d_in[26];
  const float* pbl  = (const float*)d_in[27];
  const float* pbr  = (const float*)d_in[28];

  const int* src  = eidx;            const int* dst  = eidx + NE;
  const int* msrc = eidx_m;          const int* mdst = eidx_m + NE;

  // -------- workspace carving --------
  float* w = (float*)d_ws; size_t off = 0;
  auto carve = [&](size_t n) { float* p = w + off; off += n; return p; };
  float* bufA    = carve((size_t)NN * 1024);   // xl (all layers)
  float* bufB    = carve((size_t)NN * 1024);   // xr (all layers)
  float* bufC    = carve((size_t)NN * 1024);   // conv1 acc -> h1
  float* bufD    = carve((size_t)NN * 256);    // conv2 acc -> h2
  float* bufE    = carve((size_t)NN * 256);    // pool  acc -> h3
  float* mean_ea = carve((size_t)NN * 16);
  float* deg     = carve((size_t)NN);
  float* nmax    = carve((size_t)NN * HEADS);
  float* nsum    = carve((size_t)NN * HEADS);
  float* logits  = carve((size_t)(NE + NN) * HEADS);

  // -------- mean edge-attr (shared by conv1 & conv2) --------
  fill_kernel<<<(NN * 16 + 255) / 256, 256, 0, stream>>>(mean_ea, 0.0f, (size_t)NN * 16);
  fill_kernel<<<(NN + 255) / 256, 256, 0, stream>>>(deg, 0.0f, (size_t)NN);
  edge_mean_acc<<<(NE * 16 + 255) / 256, 256, 0, stream>>>(dst, eattr, mean_ea, deg);
  edge_mean_div<<<(NN * 16 + 255) / 256, 256, 0, stream>>>(mean_ea, deg);

  // -------- conv1: in 256 -> 4 heads x 256, relu --------
  run_gatv2(stream, x, 256, 256, c1Wl, c1bl, c1Wr, c1br, c1We, c1att, c1b,
            src, dst, eattr, mean_ea, bufA, bufB, bufC, logits, nmax, nsum, 1);
  // -------- conv2: in 1024 -> 4 heads x 64, relu --------
  run_gatv2(stream, bufC, 1024, 64, c2Wl, c2bl, c2Wr, c2br, c2We, c2att, c2b,
            src, dst, eattr, mean_ea, bufA, bufB, bufD, logits, nmax, nsum, 1);
  // -------- pool: in 256 -> 4 heads x 64, master edges, no edge_attr, no relu --------
  run_gatv2(stream, bufD, 256, 64, pWl, pbl, pWr, pbr, nullptr, patt, pb,
            msrc, mdst, nullptr, nullptr, bufA, bufB, bufE, logits, nmax, nsum, 0);

  // -------- readout + MLP head --------
  fc_kernel<<<1, NG * 64, 0, stream>>>(bufE, n_nodes, fc1_w, fc1_b, fc2_w, fc2_b,
                                       (float*)d_out);
}
